// GAT_1348619731439
// MI455X (gfx1250) — matrix operand order repaired and checked by reading, not compiled, not run on hardware
//
#include <hip/hip_runtime.h>
#include <hip/hip_bf16.h>
#include <math.h>

typedef __attribute__((ext_vector_type(16))) _Float16 v16h;
typedef __attribute__((ext_vector_type(8)))  float    v8f;

#define NNODES 100000
#define NEDGES 1600000
#define ETOT   (NEDGES + NNODES)   // self-loops appended
#define FIN    500
#define NHEADS 8
#define HID    8
#define NCLS   7
#define NEG_SLOPE 0.2f

// ---------------------------------------------------------------------------
// WMMA GEMM:  C[NNODES, ldc] = A[NNODES,K] * B[NVALID,K]^T  (row-major A, B)
// K and NVALID are compile-time so the K-tail and the N-column guard fold
// away: the steady-state loop is branch-free (EXEC untouched) and the tail
// chunk is only instantiated when K % 32 != 0.
// One wave32 computes a 16x64 row-strip (four 16x16 N-tiles): the A fragment
// is loaded once per 32-wide K-chunk and reused by 4 WMMAs.  All control flow
// is wave-uniform, so EXEC==all-1s at every v_wmma as the ISA requires
// (the only per-lane predicate is the last B fragment when NVALID==56, which
// the compiler resolves with a select/cndmask on loads, outside the WMMAs).
// ---------------------------------------------------------------------------
template <int K, int NVALID>
__global__ void gemm_wmma_f16(const float* __restrict__ A,
                              const float* __restrict__ B,
                              float* __restrict__ C,
                              int ldc, int Mt) {
  const int wave = threadIdx.x >> 5;
  const int lane = threadIdx.x & 31;
  const int tm   = blockIdx.x * (blockDim.x >> 5) + wave;  // M tile (row strip)
  if (tm >= Mt) return;                       // wave-uniform
  const int half = lane >> 4;
  const int r    = lane & 15;

  const int m = tm * 16 + r;                  // A row handled by this lane
  const float* __restrict__ arow = A + (long)m * K;
  const float* browt[4];
  #pragma unroll
  for (int t = 0; t < 4; ++t) browt[t] = B + (long)(r + 16 * t) * K;

  v8f acc[4] = {{}, {}, {}, {}};

  constexpr int KFULL = (K / 32) * 32;
  for (int kc = 0; kc < KFULL; kc += 32) {    // uniform, branch-free loop
    v16h a;
    const int k1 = kc + 16 * half;            // A run 1: k1..k1+7
    const int k2 = k1 + 8;                    // A run 2: k2..k2+7
    const int kb = kc + 16 * half;            // B run  : kb..kb+15
    {
      float4 a0 = *(const float4*)(arow + k1);
      float4 a1 = *(const float4*)(arow + k1 + 4);
      float4 a2 = *(const float4*)(arow + k2);
      float4 a3 = *(const float4*)(arow + k2 + 4);
      a[0]=(_Float16)a0.x;  a[1]=(_Float16)a0.y;  a[2]=(_Float16)a0.z;  a[3]=(_Float16)a0.w;
      a[4]=(_Float16)a1.x;  a[5]=(_Float16)a1.y;  a[6]=(_Float16)a1.z;  a[7]=(_Float16)a1.w;
      a[8]=(_Float16)a2.x;  a[9]=(_Float16)a2.y;  a[10]=(_Float16)a2.z; a[11]=(_Float16)a2.w;
      a[12]=(_Float16)a3.x; a[13]=(_Float16)a3.y; a[14]=(_Float16)a3.z; a[15]=(_Float16)a3.w;
    }
    v16h bt[4];
    #pragma unroll
    for (int t = 0; t < 4; ++t) {
      if (t * 16 + 15 < NVALID) {             // compile-time: whole fragment valid
        float4 b0 = *(const float4*)(browt[t] + kb);
        float4 b1 = *(const float4*)(browt[t] + kb + 4);
        float4 b2 = *(const float4*)(browt[t] + kb + 8);
        float4 b3 = *(const float4*)(browt[t] + kb + 12);
        bt[t][0]=(_Float16)b0.x;  bt[t][1]=(_Float16)b0.y;  bt[t][2]=(_Float16)b0.z;  bt[t][3]=(_Float16)b0.w;
        bt[t][4]=(_Float16)b1.x;  bt[t][5]=(_Float16)b1.y;  bt[t][6]=(_Float16)b1.z;  bt[t][7]=(_Float16)b1.w;
        bt[t][8]=(_Float16)b2.x;  bt[t][9]=(_Float16)b2.y;  bt[t][10]=(_Float16)b2.z; bt[t][11]=(_Float16)b2.w;
        bt[t][12]=(_Float16)b3.x; bt[t][13]=(_Float16)b3.y; bt[t][14]=(_Float16)b3.z; bt[t][15]=(_Float16)b3.w;
      } else {                                // compile-time: partial fragment
        bool bv = (t * 16 + r < NVALID);      // per-lane select, no EXEC change
        #pragma unroll
        for (int i = 0; i < 16; ++i)
          bt[t][i] = (_Float16)(bv ? browt[t][kb + i] : 0.0f);
      }
    }
    acc[0] = __builtin_amdgcn_wmma_f32_16x16x32_f16(false, a, false, bt[0], (short)0, acc[0], false, false);
    acc[1] = __builtin_amdgcn_wmma_f32_16x16x32_f16(false, a, false, bt[1], (short)0, acc[1], false, false);
    acc[2] = __builtin_amdgcn_wmma_f32_16x16x32_f16(false, a, false, bt[2], (short)0, acc[2], false, false);
    acc[3] = __builtin_amdgcn_wmma_f32_16x16x32_f16(false, a, false, bt[3], (short)0, acc[3], false, false);
  }

  if (K % 32 != 0) {                          // compile-time tail chunk
    constexpr int kc = KFULL;
    v16h a;
    const int k1 = kc + 16 * half;
    const int k2 = k1 + 8;
    const int kb = kc + 16 * half;
    #pragma unroll
    for (int i = 0; i < 8; ++i) {
      int ka  = k1 + i;
      int ka2 = k2 + i;
      a[i]     = (_Float16)((ka  < K) ? arow[ka]  : 0.0f);
      a[8 + i] = (_Float16)((ka2 < K) ? arow[ka2] : 0.0f);
    }
    v16h bt[4];
    #pragma unroll
    for (int t = 0; t < 4; ++t) {
      bool bv = (t * 16 + r < NVALID);
      #pragma unroll
      for (int i = 0; i < 16; ++i) {
        int kk = kb + i;
        bt[t][i] = (_Float16)((bv && kk < K) ? browt[t][kk] : 0.0f);
      }
    }
    acc[0] = __builtin_amdgcn_wmma_f32_16x16x32_f16(false, a, false, bt[0], (short)0, acc[0], false, false);
    acc[1] = __builtin_amdgcn_wmma_f32_16x16x32_f16(false, a, false, bt[1], (short)0, acc[1], false, false);
    acc[2] = __builtin_amdgcn_wmma_f32_16x16x32_f16(false, a, false, bt[2], (short)0, acc[2], false, false);
    acc[3] = __builtin_amdgcn_wmma_f32_16x16x32_f16(false, a, false, bt[3], (short)0, acc[3], false, false);
  }

  // D layout: VGPR v -> row = tm*16 + v + 8*half, col = 16*t + (lane&15)
  #pragma unroll
  for (int v = 0; v < 8; ++v) {
    int row = tm * 16 + v + 8 * half;         // NNODES % 16 == 0: always valid
    float* crow = C + (long)row * ldc;
    #pragma unroll
    for (int t = 0; t < 4; ++t) {
      if (t * 16 + 15 < NVALID) crow[t * 16 + r] = acc[t][v];
      else if (t * 16 + r < NVALID) crow[t * 16 + r] = acc[t][v];
    }
  }
}

// ---------------------------------------------------------------------------
// alpha_s / alpha_d:  per (node, head) dot of h[n,head,:] with att vectors
// ---------------------------------------------------------------------------
__global__ void alpha_kernel(const float* __restrict__ h,
                             const float* __restrict__ asrc,
                             const float* __restrict__ adst,
                             float* __restrict__ out_s,
                             float* __restrict__ out_d, int D) {
  int idx = blockIdx.x * blockDim.x + threadIdx.x;
  if (idx >= NNODES * NHEADS) return;
  int n = idx / NHEADS, hh = idx % NHEADS;
  const float* base = h + (long)n * (NHEADS * D) + hh * D;
  const float* as = asrc + hh * D;
  const float* ad = adst + hh * D;
  float s = 0.f, d = 0.f;
  for (int j = 0; j < D; ++j) { float v = base[j]; s += v * as[j]; d += v * ad[j]; }
  out_s[idx] = s;
  out_d[idx] = d;
}

__global__ void fill_kernel(float* __restrict__ p, float v, int n) {
  int i = blockIdx.x * blockDim.x + threadIdx.x;
  if (i < n) p[i] = v;
}

__device__ __forceinline__ void edge_nodes(const int* __restrict__ ei, int e,
                                           int& s, int& d) {
  if (e < NEDGES) { s = ei[e]; d = ei[NEDGES + e]; }
  else            { s = e - NEDGES; d = s; }           // appended self-loop
}

__device__ __forceinline__ float lrelu(float x) {
  return x > 0.f ? x : NEG_SLOPE * x;
}

// IEEE-ordered float atomic max via sign-split int/uint atomics
__device__ __forceinline__ void atomicMaxFloat(float* addr, float val) {
  if (val >= 0.f) atomicMax((int*)addr, __float_as_int(val));
  else            atomicMin((unsigned int*)addr, __float_as_uint(val));
}

__global__ void edge_max_kernel(const int* __restrict__ ei,
                                const float* __restrict__ as,
                                const float* __restrict__ ad,
                                float* __restrict__ emax) {
  int idx = blockIdx.x * blockDim.x + threadIdx.x;
  if (idx >= ETOT * NHEADS) return;
  int e = idx / NHEADS, hh = idx % NHEADS;
  int s, d; edge_nodes(ei, e, s, d);
  float ev = lrelu(as[s * NHEADS + hh] + ad[d * NHEADS + hh]);
  atomicMaxFloat(&emax[d * NHEADS + hh], ev);
}

__global__ void edge_sum_kernel(const int* __restrict__ ei,
                                const float* __restrict__ as,
                                const float* __restrict__ ad,
                                const float* __restrict__ emax,
                                float* __restrict__ den) {
  int idx = blockIdx.x * blockDim.x + threadIdx.x;
  if (idx >= ETOT * NHEADS) return;
  int e = idx / NHEADS, hh = idx % NHEADS;
  int s, d; edge_nodes(ei, e, s, d);
  float ev = lrelu(as[s * NHEADS + hh] + ad[d * NHEADS + hh]);
  atomicAdd(&den[d * NHEADS + hh], __expf(ev - emax[d * NHEADS + hh]));
}

__global__ void edge_msg_kernel(const int* __restrict__ ei,
                                const float* __restrict__ as,
                                const float* __restrict__ ad,
                                const float* __restrict__ emax,
                                const float* __restrict__ den,
                                const float* __restrict__ hsrc,
                                float* __restrict__ agg, int D) {
  int idx = blockIdx.x * blockDim.x + threadIdx.x;
  if (idx >= ETOT * NHEADS) return;
  int e = idx / NHEADS, hh = idx % NHEADS;
  int s, d; edge_nodes(ei, e, s, d);
  float ev = lrelu(as[s * NHEADS + hh] + ad[d * NHEADS + hh]);
  float w  = __expf(ev - emax[d * NHEADS + hh]) / den[d * NHEADS + hh];
  const int hd = NHEADS * D;
  const float* hs = hsrc + (long)s * hd + hh * D;
  float*       ao = agg  + (long)d * hd + hh * D;
  for (int j = 0; j < D; ++j) atomicAdd(&ao[j], hs[j] * w);
}

// agg = elu(agg + bias)  (in place; becomes layer-2 input)
__global__ void bias_elu_kernel(float* __restrict__ agg,
                                const float* __restrict__ bias, int n, int ld) {
  int i = blockIdx.x * blockDim.x + threadIdx.x;
  if (i >= n) return;
  float v = agg[i] + bias[i % ld];
  agg[i] = v > 0.f ? v : (__expf(v) - 1.0f);
}

__global__ void bias_add_kernel(float* __restrict__ out,
                                const float* __restrict__ bias, int n, int ld) {
  int i = blockIdx.x * blockDim.x + threadIdx.x;
  if (i >= n) return;
  out[i] += bias[i % ld];
}

static inline int cdiv(long a, long b) { return (int)((a + b - 1) / b); }

extern "C" void kernel_launch(void* const* d_in, const int* in_sizes, int n_in,
                              void* d_out, int out_size, void* d_ws, size_t ws_size,
                              hipStream_t stream) {
  const float* x    = (const float*)d_in[0];
  const int*   ei   = (const int*)  d_in[1];
  const float* W1   = (const float*)d_in[2];
  const float* as1  = (const float*)d_in[3];
  const float* ad1  = (const float*)d_in[4];
  const float* b1   = (const float*)d_in[5];
  const float* W2   = (const float*)d_in[6];
  const float* as2  = (const float*)d_in[7];
  const float* ad2  = (const float*)d_in[8];
  const float* b2   = (const float*)d_in[9];
  float* out = (float*)d_out;

  // workspace layout (floats)
  float* ws   = (float*)d_ws;
  float* H    = ws;                          // [N,64] layer-1 features, reused as [N,56] layer-2 features
  float* AGG  = H   + (long)NNODES * 64;     // [N,64] aggregation -> elu -> layer-2 input
  float* AS   = AGG + (long)NNODES * 64;     // [N,8]
  float* AD   = AS  + (long)NNODES * NHEADS; // [N,8]
  float* EMAX = AD  + (long)NNODES * NHEADS; // [N,8]
  float* DEN  = EMAX+ (long)NNODES * NHEADS; // [N,8]

  const int NH     = NNODES * NHEADS;
  const int EH     = ETOT * NHEADS;
  const int nh_blk = cdiv(NH, 256);
  const int eh_blk = cdiv(EH, 256);
  const int Mt     = NNODES / 16;            // 6250 row-strips, each wave does 16x64

  // ---------------- layer 1 ----------------
  gemm_wmma_f16<FIN, 64><<<cdiv(Mt, 8), 256, 0, stream>>>(x, W1, H, 64, Mt);
  alpha_kernel<<<nh_blk, 256, 0, stream>>>(H, as1, ad1, AS, AD, HID);
  fill_kernel<<<nh_blk, 256, 0, stream>>>(EMAX, -3.402823466e38f, NH);
  hipMemsetAsync(DEN, 0, (size_t)NH * sizeof(float), stream);
  hipMemsetAsync(AGG, 0, (size_t)NNODES * 64 * sizeof(float), stream);
  edge_max_kernel<<<eh_blk, 256, 0, stream>>>(ei, AS, AD, EMAX);
  edge_sum_kernel<<<eh_blk, 256, 0, stream>>>(ei, AS, AD, EMAX, DEN);
  edge_msg_kernel<<<eh_blk, 256, 0, stream>>>(ei, AS, AD, EMAX, DEN, H, AGG, HID);
  bias_elu_kernel<<<cdiv((long)NNODES * 64, 256), 256, 0, stream>>>(
      AGG, b1, NNODES * 64, 64);

  // ---------------- layer 2 ----------------
  gemm_wmma_f16<64, NHEADS * NCLS><<<cdiv(Mt, 8), 256, 0, stream>>>(
      AGG, W2, H, NHEADS * NCLS, Mt);
  alpha_kernel<<<nh_blk, 256, 0, stream>>>(H, as2, ad2, AS, AD, NCLS);
  fill_kernel<<<nh_blk, 256, 0, stream>>>(EMAX, -3.402823466e38f, NH);
  hipMemsetAsync(DEN, 0, (size_t)NH * sizeof(float), stream);
  hipMemsetAsync(out, 0, (size_t)NNODES * NHEADS * NCLS * sizeof(float), stream);
  edge_max_kernel<<<eh_blk, 256, 0, stream>>>(ei, AS, AD, EMAX);
  edge_sum_kernel<<<eh_blk, 256, 0, stream>>>(ei, AS, AD, EMAX, DEN);
  edge_msg_kernel<<<eh_blk, 256, 0, stream>>>(ei, AS, AD, EMAX, DEN, H, out, NCLS);
  bias_add_kernel<<<cdiv((long)NNODES * NHEADS * NCLS, 256), 256, 0, stream>>>(
      out, b2, NNODES * NHEADS * NCLS, NHEADS * NCLS);
}
